// CosineSimCodebook_24189255811229
// MI455X (gfx1250) — matrix operand order repaired and checked by reading, not compiled, not run on hardware
//
#include <hip/hip_runtime.h>
#include <hip/hip_bf16.h>

// Problem constants (from reference setup_inputs):
// x:      [4, 4096, 512]  f32  -> M = 16384, K = 512
// embed:  [1, 8192, 512]  f32  -> C = 8192 codes, D = 512
#define MDIM 16384
#define CDIM 8192
#define KDIM 512

// GEMM tiling
#define BM 128
#define BN 128
#define BK 32                  // one bf16 WMMA k-step
#define LDS_STRIDE (BK + 8)    // TDM pads each 64B row by 16B -> stride 40 bf16, conflict-free

typedef __attribute__((ext_vector_type(16))) __bf16 v16bf;
typedef __attribute__((ext_vector_type(8)))  __bf16 bf16x8;
typedef __attribute__((ext_vector_type(8)))  float  v8f;
typedef __attribute__((ext_vector_type(4)))  unsigned int u32x4;
typedef __attribute__((ext_vector_type(8)))  int i32x8;
typedef __attribute__((ext_vector_type(4)))  int i32x4;

union FragU { bf16x8 h[2]; v16bf v; };

// LDS offset of a __shared__ object: addrspacecast generic->local strips the aperture.
__device__ __forceinline__ unsigned lds_offset_of(const void* p) {
    return (unsigned)(unsigned long long)
        (__attribute__((address_space(3))) const char*)p;
}

// Issue one TDM descriptor: load a BM x BK bf16 tile (row-major, row stride KDIM)
// from global into LDS with 16B padding appended after every 64B row.
__device__ __forceinline__ void tdm_load_tile(const __bf16* gptr, unsigned lds_byte_off,
                                              unsigned tensor_rows) {
    const unsigned long long ga = (unsigned long long)gptr;
    u32x4 g0;
    g0[0] = 1u;                                            // count=1 (user descriptor)
    g0[1] = lds_byte_off;                                  // lds_addr
    g0[2] = (unsigned)(ga & 0xFFFFFFFFu);                  // global_addr[31:0]
    g0[3] = (unsigned)((ga >> 32) & 0x01FFFFFFu)           // global_addr[56:32]
          | (2u << 30);                                    // type=2 ("image")
    i32x8 g1;
    g1[0] = (1 << 16)                                      // data_size = 2 bytes
          | (1 << 20)                                      // pad_enable
          | (3 << 22)                                      // pad_interval: 16 DWORDs (64B row)
          | (3 << 25);                                     // pad_amount:   4 DWORDs (16B)
    g1[1] = (int)((KDIM & 0xFFFFu) << 16);                 // tensor_dim0[15:0]   (=512)
    g1[2] = (int)(((unsigned)KDIM >> 16) |
                  ((tensor_rows & 0xFFFFu) << 16));        // dim0[31:16] | dim1[15:0]
    g1[3] = (int)(((tensor_rows >> 16) & 0xFFFFu) |
                  ((unsigned)BK << 16));                   // dim1[31:16] | tile_dim0 = 32
    g1[4] = BM;                                            // tile_dim1 = 128 (tile_dim2 = 0)
    g1[5] = KDIM;                                          // tensor_dim0_stride[31:0] = 512
    g1[6] = 0;                                             // dim0_stride[47:32] | dim1_stride lo
    g1[7] = 0;
    const i32x4 z4 = {0, 0, 0, 0};
#if defined(__clang_major__) && (__clang_major__ >= 23)
    const i32x8 z8 = {0, 0, 0, 0, 0, 0, 0, 0};
    __builtin_amdgcn_tensor_load_to_lds(g0, g1, z4, z4, z8, 0);
#else
    __builtin_amdgcn_tensor_load_to_lds(g0, g1, z4, z4, 0);
#endif
}

// ---------------------------------------------------------------------------
// Kernel 1: pack x and embed from f32 to bf16 (8 elements / thread, b128 out)
// ---------------------------------------------------------------------------
__global__ __launch_bounds__(256) void pack_bf16_kernel(const float* __restrict__ x,
                                                        const float* __restrict__ embed,
                                                        __bf16* __restrict__ xbf,
                                                        __bf16* __restrict__ ebf) {
    const int idx = blockIdx.x * 256 + threadIdx.x;   // one thread = 8 elements
    const int x8 = (MDIM * KDIM) / 8;                 // 1,048,576
    const float* src;
    __bf16* dst;
    if (idx < x8) {
        src = x + (size_t)idx * 8;
        dst = xbf + (size_t)idx * 8;
    } else {
        const int j = idx - x8;
        src = embed + (size_t)j * 8;
        dst = ebf + (size_t)j * 8;
    }
    const float4 a = ((const float4*)src)[0];
    const float4 b = ((const float4*)src)[1];
    bf16x8 o;
    o[0] = (__bf16)a.x; o[1] = (__bf16)a.y; o[2] = (__bf16)a.z; o[3] = (__bf16)a.w;
    o[4] = (__bf16)b.x; o[5] = (__bf16)b.y; o[6] = (__bf16)b.z; o[7] = (__bf16)b.w;
    *(bf16x8*)dst = o;
}

// ---------------------------------------------------------------------------
// Kernel 2: zero the per-row packed (max,argmax) table
// ---------------------------------------------------------------------------
__global__ __launch_bounds__(256) void init_rowmax_kernel(unsigned long long* __restrict__ rowmax) {
    const int i = blockIdx.x * 256 + threadIdx.x;
    if (i < MDIM) rowmax[i] = 0ull;
}

// ---------------------------------------------------------------------------
// Kernel 3: bf16 WMMA GEMM  dist[m,c] = sum_k x[m,k]*embed[c,k]
//   block tile 128x128, 8 waves (wave32): wave grid 4 (M) x 2 (N),
//   each wave: 2x4 grid of 16x16 WMMA tiles (32 rows x 64 cols).
//   A/B tiles streamed by the Tensor Data Mover into double-buffered LDS;
//   wave 0 issues descriptors, TENSORcnt + workgroup barrier synchronize.
//   Epilogue: store dist (f32) + fused per-row argmax via packed u64 atomicMax.
// ---------------------------------------------------------------------------
__global__ __launch_bounds__(256) void gemm_argmax_kernel(const __bf16* __restrict__ xbf,
                                                          const __bf16* __restrict__ ebf,
                                                          float* __restrict__ dist,
                                                          unsigned long long* __restrict__ rowmax) {
    // double-buffered: [buf][ A: BM rows | B: BN rows ] with padded stride
    __shared__ __bf16 smem[2 * (BM + BN) * LDS_STRIDE];

    const int tid    = threadIdx.x;
    const int lane   = tid & 31;
    const int wid    = tid >> 5;          // 0..7
    const int wave_m = wid & 3;           // 4 waves along M
    const int wave_n = wid >> 2;          // 2 waves along N
    const int block_m = blockIdx.y * BM;
    const int block_n = blockIdx.x * BN;

    const int half   = lane >> 4;         // 0: lanes 0-15, 1: lanes 16-31
    const int lane15 = lane & 15;

    const unsigned buf_elems = (BM + BN) * LDS_STRIDE;
    const unsigned lds_base  = lds_offset_of(smem);

    v8f acc[2][4];
#pragma unroll
    for (int mt = 0; mt < 2; ++mt)
#pragma unroll
        for (int nt = 0; nt < 4; ++nt)
            acc[mt][nt] = (v8f){0.f, 0.f, 0.f, 0.f, 0.f, 0.f, 0.f, 0.f};

    const int nK = KDIM / BK;             // 16 k-steps

    // prologue: DMA tile 0 into buffer 0
    if (wid == 0) {
        tdm_load_tile(xbf + (size_t)block_m * KDIM, lds_base, MDIM);
        tdm_load_tile(ebf + (size_t)block_n * KDIM,
                      lds_base + BM * LDS_STRIDE * 2, CDIM);
    }

    for (int i = 0; i < nK; ++i) {
        if (wid == 0) {
            if (i + 1 < nK) {
                // DMA next tile into the other buffer, then wait for the
                // oldest pair (tile i) while tile i+1 stays in flight.
                const unsigned nb = (unsigned)((i + 1) & 1) * buf_elems * 2;
                const int kt = (i + 1) * BK;
                tdm_load_tile(xbf + (size_t)block_m * KDIM + kt, lds_base + nb, MDIM);
                tdm_load_tile(ebf + (size_t)block_n * KDIM + kt,
                              lds_base + nb + BM * LDS_STRIDE * 2, CDIM);
                __builtin_amdgcn_s_wait_tensorcnt(2);
            } else {
                __builtin_amdgcn_s_wait_tensorcnt(0);
            }
        }
        __syncthreads();                  // tile i visible to all waves

        const __bf16* lds_a = smem + (size_t)(i & 1) * buf_elems;
        const __bf16* lds_b = lds_a + BM * LDS_STRIDE;

        // ---- load WMMA fragments from LDS ----
        // A (16x32 bf16): lane row = M, lanes 0-15 hold K {0..7,16..23}, lanes 16-31 {8..15,24..31}
        FragU afrag[2];
#pragma unroll
        for (int mt = 0; mt < 2; ++mt) {
            const __bf16* base = lds_a + (wave_m * 32 + mt * 16 + lane15) * LDS_STRIDE;
            afrag[mt].h[0] = *(const bf16x8*)(base + half * 8);
            afrag[mt].h[1] = *(const bf16x8*)(base + 16 + half * 8);
        }
        // B (32x16 bf16): lane col = N, lanes 0-15 hold K {0..7,16..23}, lanes 16-31 {8..15,24..31}
        FragU bfrag[4];
#pragma unroll
        for (int nt = 0; nt < 4; ++nt) {
            const __bf16* base = lds_b + (wave_n * 64 + nt * 16 + lane15) * LDS_STRIDE;
            bfrag[nt].h[0] = *(const bf16x8*)(base + half * 8);
            bfrag[nt].h[1] = *(const bf16x8*)(base + 16 + half * 8);
        }

        // ---- 8 WMMAs ----
#pragma unroll
        for (int mt = 0; mt < 2; ++mt)
#pragma unroll
            for (int nt = 0; nt < 4; ++nt)
                acc[mt][nt] = __builtin_amdgcn_wmma_f32_16x16x32_bf16(
                    false, afrag[mt].v, false, bfrag[nt].v,
                    (short)0, acc[mt][nt], false, false);

        __syncthreads();                  // all reads of buffer (i&1) done
    }

    // ---- epilogue: store dist + fused per-row argmax ----
    // C/D layout: reg r -> row r (lanes 0-15) / row r+8 (lanes 16-31), col = lane&15.
#pragma unroll
    for (int mt = 0; mt < 2; ++mt) {
#pragma unroll
        for (int r = 0; r < 8; ++r) {
            const int m = block_m + wave_m * 32 + mt * 16 + r + half * 8;

            // store this row-slice of dist (4 column tiles)
#pragma unroll
            for (int nt = 0; nt < 4; ++nt) {
                const int c = block_n + wave_n * 64 + nt * 16 + lane15;
                dist[(size_t)m * CDIM + c] = acc[mt][nt][r];
            }

            // local max across the 4 column tiles
            float    best_v = acc[mt][0][r];
            unsigned best_c = (unsigned)(block_n + wave_n * 64 + lane15);
#pragma unroll
            for (int nt = 1; nt < 4; ++nt) {
                const float    v = acc[mt][nt][r];
                const unsigned c = (unsigned)(block_n + wave_n * 64 + nt * 16 + lane15);
                if (v > best_v || (v == best_v && c < best_c)) { best_v = v; best_c = c; }
            }
            // 16-lane butterfly: masks 1,2,4,8 stay within each half (rows differ per half)
#pragma unroll
            for (int off = 1; off < 16; off <<= 1) {
                const float    vo = __shfl_xor(best_v, off, 32);
                const unsigned co = __shfl_xor(best_c, off, 32);
                if (vo > best_v || (vo == best_v && co < best_c)) { best_v = vo; best_c = co; }
            }
            if (lane15 == 0) {
                unsigned vb = __float_as_uint(best_v);
                vb = (vb & 0x80000000u) ? ~vb : (vb | 0x80000000u);  // order-preserving encode
                const unsigned long long packed =
                    ((unsigned long long)vb << 32) | (unsigned long long)(~best_c);
                atomicMax(rowmax + m, packed);
            }
        }
    }
}

// ---------------------------------------------------------------------------
// Kernel 4: decode argmax, emit embed_ind, gather quantize rows (float4)
// ---------------------------------------------------------------------------
__global__ __launch_bounds__(128) void finalize_kernel(const unsigned long long* __restrict__ rowmax,
                                                       const float* __restrict__ embed,
                                                       float* __restrict__ q_out,
                                                       float* __restrict__ ind_out) {
    const int row = blockIdx.x;
    const unsigned long long packed = rowmax[row];
    const unsigned c = ~((unsigned)packed);
    if (threadIdx.x == 0) ind_out[row] = (float)c;
    const float4* src = (const float4*)embed + (size_t)c * (KDIM / 4);
    float4*       dst = (float4*)q_out + (size_t)row * (KDIM / 4);
    dst[threadIdx.x] = src[threadIdx.x];   // 128 threads x float4 = 512 floats
}

// ---------------------------------------------------------------------------
extern "C" void kernel_launch(void* const* d_in, const int* in_sizes, int n_in,
                              void* d_out, int out_size, void* d_ws, size_t ws_size,
                              hipStream_t stream) {
    const float* x     = (const float*)d_in[0];   // [4,4096,512]
    const float* embed = (const float*)d_in[1];   // [1,8192,512]

    // workspace: x_bf16 (16 MB) | embed_bf16 (8 MB) | rowmax u64 (128 KB)
    char* ws = (char*)d_ws;
    __bf16* xbf = (__bf16*)ws;
    __bf16* ebf = (__bf16*)(ws + (size_t)MDIM * KDIM * 2);
    unsigned long long* rowmax =
        (unsigned long long*)(ws + (size_t)(MDIM + CDIM) * KDIM * 2);

    // output: quantize [16384,512] | embed_ind [16384] | dist [16384,8192]
    float* q_out    = (float*)d_out;
    float* ind_out  = q_out + (size_t)MDIM * KDIM;
    float* dist_out = ind_out + MDIM;

    const int pack_blocks = ((MDIM + CDIM) * KDIM / 8) / 256;       // 6144
    pack_bf16_kernel<<<pack_blocks, 256, 0, stream>>>(x, embed, xbf, ebf);

    init_rowmax_kernel<<<(MDIM + 255) / 256, 256, 0, stream>>>(rowmax);

    dim3 grid(CDIM / BN, MDIM / BM);                                 // 64 x 128
    gemm_argmax_kernel<<<grid, 256, 0, stream>>>(xbf, ebf, dist_out, rowmax);

    finalize_kernel<<<MDIM, 128, 0, stream>>>(rowmax, embed, q_out, ind_out);
}
